// CIFlow_56435870269615
// MI455X (gfx1250) — matrix operand order repaired and checked by reading, not compiled, not run on hardware
//
#include <hip/hip_runtime.h>
#include <hip/hip_bf16.h>
#include <stdint.h>

// ---------------------------------------------------------------------------
// CIFlow pipeline for gfx1250 (MI455X). Dense GEMMs use V_WMMA_F32_16X16X32_BF16
// (wave32, 16x16 tiles, f32 accumulate). Sparse/elementwise stages use VALU +
// global f32 atomics (L2-resident working set, 192MB L2).
// ---------------------------------------------------------------------------

#define N_NODES 65536
#define N_EDGES 1048576
#define N_GRAPH 512
#define DIM     128      // D_IN == D1 == D2 == 128
#define DHID    256      // 2*D2
#define KCL     10
#define NFILT   16
#define NCLS    10
#define NLAYER  3

typedef __attribute__((ext_vector_type(16))) __bf16 v16bf;
typedef __attribute__((ext_vector_type(8)))  __bf16 v8bf;
typedef __attribute__((ext_vector_type(8)))  float  v8f;

__device__ __forceinline__ __bf16 f2bf(float x) { return (__bf16)x; }

// --- WMMA fragment loaders -------------------------------------------------
// A fragment (16x32 bf16, M x K): lane l: m=l&15, half=l>>4.
//   elems 0..7  -> K = kt + half*8 + 0..7
//   elems 8..15 -> K = kt + 16 + half*8 + 0..7
__device__ __forceinline__ v16bf load_a16(const __bf16* base, int stride, int kt, int lane) {
    int m = lane & 15, half = lane >> 4;
    const __bf16* p = base + m * stride + kt + half * 8;
    v8bf lo = *(const v8bf*)p;
    v8bf hi = *(const v8bf*)(p + 16);
    v16bf r;
#pragma unroll
    for (int i = 0; i < 8; ++i) { r[i] = lo[i]; r[i + 8] = hi[i]; }
    return r;
}

// B fragment (32x16 bf16, K x N) from K-panel-packed weights:
//   pack layout: [K/32][Ncols][32], elem (kp,n,kk) = W[kp*32+kk][n]
//   lane l: n = nt + (l&15), half = l>>4, elems 0..7: K = kt + half*8 + e, elems 8..15: K = kt + 16 + half*8 + (e-8)
__device__ __forceinline__ v16bf load_b16(const __bf16* wp, int ncols, int kt, int nt, int lane) {
    int n = nt + (lane & 15), half = lane >> 4;
    const __bf16* p = wp + ((size_t)(kt >> 5) * ncols + n) * 32 + half * 8;
    v8bf lo = *(const v8bf*)p;
    v8bf hi = *(const v8bf*)(p + 16);
    v16bf r;
#pragma unroll
    for (int i = 0; i < 8; ++i) { r[i] = lo[i]; r[i + 8] = hi[i]; }
    return r;
}

// --- pack f32 [K,N] row-major -> bf16 K-panels -----------------------------
__global__ void pack_panels(const float* __restrict__ src, __bf16* __restrict__ dst,
                            int K, int Ncols) {
    int idx = blockIdx.x * blockDim.x + threadIdx.x;
    if (idx >= K * Ncols) return;
    int k = idx / Ncols, n = idx % Ncols;
    dst[((size_t)(k >> 5) * Ncols + n) * 32 + (k & 31)] = f2bf(src[idx]);
}

// --- fused block encoder: X_b = relu(F @ W1_b + b1_b) @ W2_b + b2_b --------
__global__ __launch_bounds__(256) void encoder_kernel(
    const float* __restrict__ feat, const __bf16* __restrict__ w1p,
    const float* __restrict__ b1, const __bf16* __restrict__ w2p,
    const float* __restrict__ b2, float* __restrict__ Xout) {
    __shared__ __align__(16) __bf16 Abuf[16 * DIM];
    __shared__ __align__(16) __bf16 Hbuf[16 * DHID];
    const int blk = blockIdx.y;
    const int row0 = blockIdx.x * 16;
    const int tid = threadIdx.x;
    for (int i = tid; i < 16 * DIM; i += 256)
        Abuf[i] = f2bf(feat[(size_t)(row0 + (i >> 7)) * DIM + (i & 127)]);
    __syncthreads();
    const int wave = tid >> 5, lane = tid & 31, half = lane >> 4;
    const __bf16* w1b = w1p + (size_t)blk * DIM * DHID;
    const float* bias1 = b1 + blk * DHID;
    // GEMM1: 16 output col-tiles over DHID=256; each wave does 2
#pragma unroll
    for (int tix = 0; tix < 2; ++tix) {
        int nt = (wave * 2 + tix) * 16;
        int n = nt + (lane & 15);
        v8f acc; float bv = bias1[n];
#pragma unroll
        for (int r = 0; r < 8; ++r) acc[r] = bv;
#pragma unroll
        for (int kt = 0; kt < DIM; kt += 32) {
            v16bf a = load_a16(Abuf, DIM, kt, lane);
            v16bf b = load_b16(w1b, DHID, kt, nt, lane);
            acc = __builtin_amdgcn_wmma_f32_16x16x32_bf16(false, a, false, b,
                                                          (short)0, acc, false, false);
        }
#pragma unroll
        for (int r = 0; r < 8; ++r) {
            float v = acc[r]; v = v > 0.f ? v : 0.f;
            Hbuf[(r + 8 * half) * DHID + n] = f2bf(v);
        }
    }
    __syncthreads();
    // GEMM2: 8 output col-tiles over DIM=128; one per wave, K=256
    {
        int nt = wave * 16;
        int n = nt + (lane & 15);
        const __bf16* w2b = w2p + (size_t)blk * DHID * DIM;
        const float* bias2 = b2 + blk * DIM;
        v8f acc; float bv = bias2[n];
#pragma unroll
        for (int r = 0; r < 8; ++r) acc[r] = bv;
#pragma unroll
        for (int kt = 0; kt < DHID; kt += 32) {
            v16bf a = load_a16(Hbuf, DHID, kt, lane);
            v16bf b = load_b16(w2b, DIM, kt, nt, lane);
            acc = __builtin_amdgcn_wmma_f32_16x16x32_bf16(false, a, false, b,
                                                          (short)0, acc, false, false);
        }
        float* Xb = Xout + (size_t)blk * N_NODES * DIM;
#pragma unroll
        for (int r = 0; r < 8; ++r)
            Xb[(size_t)(row0 + r + 8 * half) * DIM + n] = acc[r];
    }
}

// --- generic WMMA GEMM: out[M,128] = (relu?)(A[M,128] @ Wpack + bias) ------
__global__ __launch_bounds__(256) void wmma_gemm128(
    const float* __restrict__ A, const __bf16* __restrict__ Wp,
    const float* __restrict__ bias, float* __restrict__ out, int relu) {
    __shared__ __align__(16) __bf16 Abuf[16 * DIM];
    const int row0 = blockIdx.x * 16;
    const int tid = threadIdx.x;
    for (int i = tid; i < 16 * DIM; i += 256)
        Abuf[i] = f2bf(A[(size_t)(row0 + (i >> 7)) * DIM + (i & 127)]);
    __syncthreads();
    const int wave = tid >> 5, lane = tid & 31, half = lane >> 4;
    int nt = wave * 16, n = nt + (lane & 15);
    v8f acc; float bv = bias[n];
#pragma unroll
    for (int r = 0; r < 8; ++r) acc[r] = bv;
#pragma unroll
    for (int kt = 0; kt < DIM; kt += 32) {
        v16bf a = load_a16(Abuf, DIM, kt, lane);
        v16bf b = load_b16(Wp, DIM, kt, nt, lane);
        acc = __builtin_amdgcn_wmma_f32_16x16x32_bf16(false, a, false, b,
                                                      (short)0, acc, false, false);
    }
#pragma unroll
    for (int r = 0; r < 8; ++r) {
        float v = acc[r];
        if (relu) v = fmaxf(v, 0.f);
        out[(size_t)(row0 + r + 8 * half) * DIM + n] = v;
    }
}

// --- hidden_rep[0] = mean over blocks of X_blocks (stored bf16) ------------
__global__ void mean_blocks(const float* __restrict__ X, __bf16* __restrict__ hid0) {
    size_t i = (size_t)blockIdx.x * blockDim.x + threadIdx.x;
    hid0[i] = f2bf(0.5f * (X[i] + X[i + (size_t)N_NODES * DIM]));
}

// --- SpMM: Y[row] += val * X[col], 32 lanes x float4 cover 128 channels ----
__global__ void spmm_kernel(const int* __restrict__ rows, const int* __restrict__ cols,
                            const float* __restrict__ vals, const float* __restrict__ X,
                            float* __restrict__ Y) {
    int t = blockIdx.x * blockDim.x + threadIdx.x;
    int e = t >> 5;
    if (e >= N_EDGES) return;
    int dv = (t & 31) * 4;
    int r = rows[e], c = cols[e];
    float w = vals[e];
    float4 x = *(const float4*)&X[(size_t)c * DIM + dv];
    float* yp = &Y[(size_t)r * DIM + dv];
    atomicAdd(yp + 0, w * x.x); atomicAdd(yp + 1, w * x.y);
    atomicAdd(yp + 2, w * x.z); atomicAdd(yp + 3, w * x.w);
}

// --- per-channel sum / sumsq over 65536 rows -------------------------------
__global__ __launch_bounds__(256) void bn_stats(const float* __restrict__ X,
                                                float* __restrict__ stats) {
    int ch = threadIdx.x & 127, rh = threadIdx.x >> 7;
    int base = blockIdx.x * 256;
    float s = 0.f, s2 = 0.f;
    for (int r = rh; r < 256; r += 2) {
        float x = X[(size_t)(base + r) * DIM + ch];
        s += x; s2 += x * x;
    }
    __shared__ float ls[256], ls2[256];
    ls[threadIdx.x] = s; ls2[threadIdx.x] = s2;
    __syncthreads();
    if (rh == 0) {
        atomicAdd(&stats[ch],       s  + ls[128 + ch]);
        atomicAdd(&stats[128 + ch], s2 + ls2[128 + ch]);
    }
}

// --- BN + relu; mode0: accumulate 0.5*y into bf16 hid; mode1: write f32 ----
__global__ void bn_apply(const float* __restrict__ X, const float* __restrict__ stats,
                         const float* __restrict__ gamma, const float* __restrict__ beta,
                         __bf16* hid, float* outf, int mode, int first) {
    size_t i = (size_t)blockIdx.x * blockDim.x + threadIdx.x;
    int ch = i & 127;
    const float invN = 1.0f / (float)N_NODES;
    float m = stats[ch] * invN;
    float v = stats[128 + ch] * invN - m * m;
    float rs = rsqrtf(v + 1e-5f);
    float y = fmaxf((X[i] - m) * rs * gamma[ch] + beta[ch], 0.f);
    if (mode == 0) {
        float acc = first ? 0.f : (float)hid[i];
        hid[i] = f2bf(acc + 0.5f * y);
    } else {
        outf[i] = y;
    }
}

// --- concat MLP: per (n,d), 4 -> 128 -> 1 ----------------------------------
__global__ __launch_bounds__(256) void concat_kernel(
    const __bf16* __restrict__ hid, const float* __restrict__ w1,
    const float* __restrict__ b1, const float* __restrict__ w2,
    const float* __restrict__ b2, float* __restrict__ out) {
    __shared__ float W1s[4 * 128], W2s[128], B1s[128];
    int tid = threadIdx.x;
    for (int i = tid; i < 512; i += 256) W1s[i] = w1[i];
    if (tid < 128) { W2s[tid] = w2[tid]; B1s[tid] = b1[tid]; }
    __syncthreads();
    size_t i = (size_t)blockIdx.x * 256 + tid;
    const size_t ND = (size_t)N_NODES * DIM;
    float x0 = (float)hid[i], x1 = (float)hid[i + ND];
    float x2 = (float)hid[i + 2 * ND], x3 = (float)hid[i + 3 * ND];
    float acc = b2[0];
#pragma unroll 4
    for (int h = 0; h < 128; ++h) {
        float a = fmaf(x0, W1s[h], fmaf(x1, W1s[128 + h],
                  fmaf(x2, W1s[256 + h], fmaf(x3, W1s[384 + h], B1s[h]))));
        acc = fmaf(fmaxf(a, 0.f), W2s[h], acc);
    }
    out[i] = acc;
}

// --- S = softmax(t1 @ smlp_w2 + b2, axis=K) --------------------------------
__global__ void smlp2_softmax(const float* __restrict__ t1, const float* __restrict__ w2,
                              const float* __restrict__ b2, float* __restrict__ S) {
    int n = blockIdx.x * blockDim.x + threadIdx.x;
    if (n >= N_NODES) return;
    float lg[KCL];
#pragma unroll
    for (int c = 0; c < KCL; ++c) lg[c] = b2[c];
    const float* x = &t1[(size_t)n * DIM];
    for (int k = 0; k < DIM; ++k) {
        float xv = x[k];
        const float* wr = &w2[k * KCL];
#pragma unroll
        for (int c = 0; c < KCL; ++c) lg[c] = fmaf(xv, wr[c], lg[c]);
    }
    float mx = lg[0];
#pragma unroll
    for (int c = 1; c < KCL; ++c) mx = fmaxf(mx, lg[c]);
    float s = 0.f;
#pragma unroll
    for (int c = 0; c < KCL; ++c) { lg[c] = expf(lg[c] - mx); s += lg[c]; }
    float inv = 1.f / s;
#pragma unroll
    for (int c = 0; c < KCL; ++c) S[(size_t)n * KCL + c] = lg[c] * inv;
}

// --- graph segment bounds (graph_indicator is sorted) ----------------------
__global__ void bounds_init(int* starts, int* ends) {
    int g = blockIdx.x * blockDim.x + threadIdx.x;
    if (g < N_GRAPH) { starts[g] = 1 << 30; ends[g] = 0; }
}
__global__ void bounds_kernel(const int* __restrict__ gi, int* starts, int* ends) {
    int n = blockIdx.x * blockDim.x + threadIdx.x;
    if (n >= N_NODES) return;
    int g = gi[n];
    atomicMin(&starts[g], n);
    atomicMax(&ends[g], n + 1);
}

// --- E[g,k,d] = sum_{n in g} S[n,k]*H[n,d]; block per graph ----------------
__global__ __launch_bounds__(256) void pool_kernel(
    const float* __restrict__ S, const float* __restrict__ H,
    const int* __restrict__ starts, const int* __restrict__ ends,
    float* __restrict__ E) {
    int g = blockIdx.x, t = threadIdx.x;
    int d = t & 127, kb = t >> 7;   // kb in {0,1}; k = kb + 2*i
    float acc[5] = {0, 0, 0, 0, 0};
    int s0 = starts[g], e0 = ends[g];
    if (s0 < e0) {
        for (int n = s0; n < e0; ++n) {
            float h = H[(size_t)n * DIM + d];
            const float* sr = &S[(size_t)n * KCL];
#pragma unroll
            for (int i = 0; i < 5; ++i) acc[i] = fmaf(sr[kb + 2 * i], h, acc[i]);
        }
    }
#pragma unroll
    for (int i = 0; i < 5; ++i)
        E[(size_t)g * (KCL * DIM) + (kb + 2 * i) * DIM + d] = acc[i];
}

// --- pred1[g,c] = Eflat[g] . final_w[:,c] + b -------------------------------
__global__ void pred1_kernel(const float* __restrict__ E, const float* __restrict__ W,
                             const float* __restrict__ b, float* __restrict__ out) {
    int t = blockIdx.x * blockDim.x + threadIdx.x;
    if (t >= N_GRAPH * NCLS) return;
    int g = t / NCLS, c = t % NCLS;
    float acc = b[c];
    const float* e = &E[(size_t)g * (KCL * DIM)];
    for (int i = 0; i < KCL * DIM; ++i) acc = fmaf(e[i], W[i * NCLS + c], acc);
    out[t] = acc;
}

// --- softmax + Gumbel-argmax sample; ind_positive; one-hot collapse --------
__global__ void sample_kernel(const float* __restrict__ pred1, const int* __restrict__ targets,
                              float* __restrict__ psm, int* __restrict__ samp,
                              float* __restrict__ ind) {
    int g = blockIdx.x * blockDim.x + threadIdx.x;
    if (g >= N_GRAPH) return;
    const float* p = &pred1[g * NCLS];
    float mx = p[0];
#pragma unroll
    for (int c = 1; c < NCLS; ++c) mx = fmaxf(mx, p[c]);
    float s = 0.f, e[NCLS];
#pragma unroll
    for (int c = 0; c < NCLS; ++c) { e[c] = expf(p[c] - mx); s += e[c]; }
    float inv = 1.f / s;
    int best = 0; float bv = -1e30f;
#pragma unroll
    for (int c = 0; c < NCLS; ++c) {
        psm[g * NCLS + c] = e[c] * inv;
        uint32_t h = (uint32_t)(g * NCLS + c) * 2654435761u + 42u;
        h ^= h >> 16; h *= 0x85ebca6bu; h ^= h >> 13; h *= 0xc2b2ae35u; h ^= h >> 16;
        float u = (float)(h >> 8) * (1.0f / 16777216.0f) + 1e-10f;
        float gl = -logf(-logf(u));
        float v = p[c] + gl;
        if (v > bv) { bv = v; best = c; }
    }
    samp[g] = best;
    ind[g] = (best == targets[g]) ? 1.0f : 0.0f;
}

// --- corr = softmax(inter, axis=0) * softmax(intra, axis=1) ----------------
__global__ void corr_kernel(const float* __restrict__ inter, const float* __restrict__ intra,
                            float* __restrict__ corr) {
    int t = threadIdx.x;
    if (t >= NCLS * NFILT) return;
    int c = t / NFILT, f = t % NFILT;
    float mx = -1e30f;
    for (int cc = 0; cc < NCLS; ++cc) mx = fmaxf(mx, inter[cc * NFILT + f]);
    float s = 0.f;
    for (int cc = 0; cc < NCLS; ++cc) s += expf(inter[cc * NFILT + f] - mx);
    float a = expf(inter[c * NFILT + f] - mx) / s;
    float mr = -1e30f;
    for (int ff = 0; ff < NFILT; ++ff) mr = fmaxf(mr, intra[c * NFILT + ff]);
    float sr = 0.f;
    for (int ff = 0; ff < NFILT; ++ff) sr += expf(intra[c * NFILT + ff] - mr);
    float b = expf(intra[c * NFILT + f] - mr) / sr;
    corr[t] = a * b;
}

// --- Q = softmax(protoh @ proto_w2 + b2, axis=NF) --------------------------
__global__ void q_kernel(const float* __restrict__ ph, const float* __restrict__ w2,
                         const float* __restrict__ b2, float* __restrict__ Q) {
    int r = blockIdx.x * blockDim.x + threadIdx.x;
    if (r >= N_GRAPH * KCL) return;
    float lg[NFILT];
#pragma unroll
    for (int f = 0; f < NFILT; ++f) lg[f] = b2[f];
    const float* x = &ph[(size_t)r * DIM];
    for (int k = 0; k < DIM; ++k) {
        float xv = x[k];
        const float* wr = &w2[k * NFILT];
#pragma unroll
        for (int f = 0; f < NFILT; ++f) lg[f] = fmaf(xv, wr[f], lg[f]);
    }
    float mx = lg[0];
#pragma unroll
    for (int f = 1; f < NFILT; ++f) mx = fmaxf(mx, lg[f]);
    float s = 0.f;
#pragma unroll
    for (int f = 0; f < NFILT; ++f) { lg[f] = expf(lg[f] - mx); s += lg[f]; }
    float inv = 1.f / s;
#pragma unroll
    for (int f = 0; f < NFILT; ++f) Q[(size_t)r * NFILT + f] = lg[f] * inv;
}

// --- feature_mask[g,k] = sum_f Q[g,k,f] * corr[cat[g],f] -------------------
__global__ void fmask_kernel(const float* __restrict__ Q, const float* __restrict__ corr,
                             const int* __restrict__ samp, float* __restrict__ fm) {
    int t = blockIdx.x * blockDim.x + threadIdx.x;
    if (t >= N_GRAPH * KCL) return;
    int g = t / KCL, cat = samp[g];
    float acc = 0.f;
    const float* q = &Q[(size_t)t * NFILT];
    const float* cr = &corr[cat * NFILT];
#pragma unroll
    for (int f = 0; f < NFILT; ++f) acc = fmaf(q[f], cr[f], acc);
    fm[t] = acc;
}

// --- pred2 ------------------------------------------------------------------
__global__ void pred2_kernel(const float* __restrict__ E, const float* __restrict__ fm,
                             const float* __restrict__ W, const float* __restrict__ b,
                             float* __restrict__ out) {
    int t = blockIdx.x * blockDim.x + threadIdx.x;
    if (t >= N_GRAPH * NCLS) return;
    int g = t / NCLS, c = t % NCLS;
    float acc = b[c];
    const float* e = &E[(size_t)g * (KCL * DIM)];
    const float* f = &fm[g * KCL];
    for (int i = 0; i < KCL * DIM; ++i)
        acc = fmaf(f[i >> 7] * e[i], W[i * NCLS + c], acc);
    out[t] = acc;
}

// ---------------------------------------------------------------------------
extern "C" void kernel_launch(void* const* d_in, const int* in_sizes, int n_in,
                              void* d_out, int out_size, void* d_ws, size_t ws_size,
                              hipStream_t stream) {
    const float* feat   = (const float*)d_in[0];
    const int*   Wrow   = (const int*)d_in[1];
    const int*   Wcol   = (const int*)d_in[2];
    const float* Wvals  = (const float*)d_in[3];
    const int*   gind   = (const int*)d_in[4];
    const int*   targ   = (const int*)d_in[5];
    const float* enc_w1 = (const float*)d_in[6];
    const float* enc_b1 = (const float*)d_in[7];
    const float* enc_w2 = (const float*)d_in[8];
    const float* enc_b2 = (const float*)d_in[9];
    const float* bn_g   = (const float*)d_in[10];
    const float* bn_b   = (const float*)d_in[11];
    const float* cw1    = (const float*)d_in[12];
    const float* cb1    = (const float*)d_in[13];
    const float* cw2    = (const float*)d_in[14];
    const float* cb2    = (const float*)d_in[15];
    const float* sw1    = (const float*)d_in[16];
    const float* sb1    = (const float*)d_in[17];
    const float* sw2    = (const float*)d_in[18];
    const float* sb2    = (const float*)d_in[19];
    const float* pw1    = (const float*)d_in[20];
    const float* pb1    = (const float*)d_in[21];
    const float* pw2    = (const float*)d_in[22];
    const float* pb2    = (const float*)d_in[23];
    const float* fw     = (const float*)d_in[24];
    const float* fb     = (const float*)d_in[25];
    const float* cinter = (const float*)d_in[26];
    const float* cintra = (const float*)d_in[27];

    float* out = (float*)d_out;
    const size_t o_pred1 = 0;
    const size_t o_pred2 = 5120;
    const size_t o_S     = 10240;
    const size_t o_H     = 665600;
    const size_t o_ind   = 9054208;
    const size_t o_E     = 9054720;
    const size_t o_Q     = 9710080;

    // workspace layout (~240 MB total)
    char* ws = (char*)d_ws;
    const size_t ND4  = (size_t)N_NODES * DIM * 4;   // 32 MB
    float*  P0     = (float*)(ws);                                  // 64 MB
    float*  P1     = (float*)(ws + 2 * ND4);                        // 64 MB
    __bf16* hid    = (__bf16*)(ws + 4 * ND4);                       // 64 MB (4 layers bf16)
    float*  t1     = (float*)(ws + 6 * ND4);                        // 32 MB
    char*   small  = ws + 7 * ND4;
    __bf16* wpk1   = (__bf16*)(small);                 small += 2 * DIM * DHID * 2;
    __bf16* wpk2   = (__bf16*)(small);                 small += 2 * DHID * DIM * 2;
    __bf16* wpkS   = (__bf16*)(small);                 small += DIM * DIM * 2;
    __bf16* wpkP   = (__bf16*)(small);                 small += DIM * DIM * 2;
    float*  stats  = (float*)(small);                  small += 1024;
    float*  protoh = (float*)(small);                  small += (size_t)N_GRAPH * KCL * DIM * 4;
    float*  psm    = (float*)(small);                  small += N_GRAPH * NCLS * 4;
    int*    samp   = (int*)(small);                    small += N_GRAPH * 4;
    float*  corrb  = (float*)(small);                  small += NCLS * NFILT * 4;
    float*  fmask  = (float*)(small);                  small += N_GRAPH * KCL * 4;
    int*    starts = (int*)(small);                    small += N_GRAPH * 4;
    int*    ends   = (int*)(small);

    // 1) pack weights to bf16 K-panels
    for (int b = 0; b < 2; ++b) {
        pack_panels<<<(DIM * DHID + 255) / 256, 256, 0, stream>>>(
            enc_w1 + (size_t)b * DIM * DHID, wpk1 + (size_t)b * DIM * DHID, DIM, DHID);
        pack_panels<<<(DHID * DIM + 255) / 256, 256, 0, stream>>>(
            enc_w2 + (size_t)b * DHID * DIM, wpk2 + (size_t)b * DHID * DIM, DHID, DIM);
    }
    pack_panels<<<(DIM * DIM + 255) / 256, 256, 0, stream>>>(sw1, wpkS, DIM, DIM);
    pack_panels<<<(DIM * DIM + 255) / 256, 256, 0, stream>>>(pw1, wpkP, DIM, DIM);

    // 2) fused block encoders (WMMA): X_blocks -> P0
    encoder_kernel<<<dim3(N_NODES / 16, 2), 256, 0, stream>>>(feat, wpk1, enc_b1,
                                                              wpk2, enc_b2, P0);
    // 3) hidden_rep[0] = mean over blocks
    mean_blocks<<<(N_NODES * DIM) / 256, 256, 0, stream>>>(P0, hid);

    // 4) propagation loop: spmm -> BN -> relu -> mean over blocks
    float* cur = P0; float* nxt = P1;
    for (int p = 0; p < NLAYER; ++p) {
        hipMemsetAsync(nxt, 0, 2 * ND4, stream);
        for (int b = 0; b < 2; ++b)
            spmm_kernel<<<(N_EDGES * 32) / 256, 256, 0, stream>>>(
                Wrow, Wcol, Wvals, cur + (size_t)b * N_NODES * DIM,
                nxt + (size_t)b * N_NODES * DIM);
        __bf16* hidp = hid + (size_t)(p + 1) * N_NODES * DIM;
        for (int b = 0; b < 2; ++b) {
            hipMemsetAsync(stats, 0, 1024, stream);
            bn_stats<<<256, 256, 0, stream>>>(nxt + (size_t)b * N_NODES * DIM, stats);
            bn_apply<<<(N_NODES * DIM) / 256, 256, 0, stream>>>(
                nxt + (size_t)b * N_NODES * DIM, stats, bn_g + p * DIM, bn_b + p * DIM,
                hidp, nullptr, 0, b == 0);
        }
        float* tmp = cur; cur = nxt; nxt = tmp;
    }

    // 5) concat MLP over layer axis, then final BN -> H_all (d_out)
    concat_kernel<<<(N_NODES * DIM) / 256, 256, 0, stream>>>(hid, cw1, cb1, cw2, cb2, t1);
    hipMemsetAsync(stats, 0, 1024, stream);
    bn_stats<<<256, 256, 0, stream>>>(t1, stats);
    bn_apply<<<(N_NODES * DIM) / 256, 256, 0, stream>>>(
        t1, stats, bn_g + NLAYER * DIM, bn_b + NLAYER * DIM, nullptr, out + o_H, 1, 0);

    // 6) S_all = softmax(relu(H @ sw1 + b1) @ sw2 + b2)  (WMMA for first GEMM)
    wmma_gemm128<<<N_NODES / 16, 256, 0, stream>>>(out + o_H, wpkS, sb1, t1, 1);
    smlp2_softmax<<<N_NODES / 256, 256, 0, stream>>>(t1, sw2, sb2, out + o_S);

    // 7) per-graph cluster pooling -> E_all (d_out)
    bounds_init<<<2, 256, 0, stream>>>(starts, ends);
    bounds_kernel<<<N_NODES / 256, 256, 0, stream>>>(gind, starts, ends);
    pool_kernel<<<N_GRAPH, 256, 0, stream>>>(out + o_S, out + o_H, starts, ends, out + o_E);

    // 8) pred1, sampling, corr
    pred1_kernel<<<(N_GRAPH * NCLS + 255) / 256, 256, 0, stream>>>(out + o_E, fw, fb,
                                                                   out + o_pred1);
    sample_kernel<<<2, 256, 0, stream>>>(out + o_pred1, targ, psm, samp, out + o_ind);
    corr_kernel<<<1, 256, 0, stream>>>(cinter, cintra, corrb);

    // 9) Q = softmax(relu(E @ pw1 + b1) @ pw2 + b2)  (WMMA for first GEMM; M=5120)
    wmma_gemm128<<<(N_GRAPH * KCL) / 16, 256, 0, stream>>>(out + o_E, wpkP, pb1, protoh, 1);
    q_kernel<<<(N_GRAPH * KCL + 255) / 256, 256, 0, stream>>>(protoh, pw2, pb2, out + o_Q);

    // 10) feature mask + pred2
    fmask_kernel<<<(N_GRAPH * KCL + 255) / 256, 256, 0, stream>>>(out + o_Q, corrb, samp, fmask);
    pred2_kernel<<<(N_GRAPH * NCLS + 255) / 256, 256, 0, stream>>>(out + o_E, fmask, fw, fb,
                                                                   out + o_pred2);
    (void)in_sizes; (void)n_in; (void)out_size; (void)ws_size; (void)psm;
}
